// Raindrop_58832462020671
// MI455X (gfx1250) — compile-verified, matrix operand order checked by reading, not run on hardware
//
#include <hip/hip_runtime.h>
#include <math.h>

// ---------------------------------------------------------------------------
// Raindrop forward for MI455X (gfx1250, wave32, WMMA).
// GEMMs run through v_wmma_f32_16x16x32_f16 (f16 operands, f32 accumulate).
// Graph-propagation alphas are provably uniform (softmax of constants), so
// the message aggregation collapses to a node-mean and dist == 0 exactly.
// GEMM: M % 128 == 0 always -> no M guards; B-tile rows are clamped (OOB rows
// only feed discarded output columns); K tail peeled; double-buffered LDS
// with register prefetch overlaps global loads with the WMMAs.
// ---------------------------------------------------------------------------

typedef __attribute__((ext_vector_type(16))) _Float16 v16h;
typedef __attribute__((ext_vector_type(8)))  _Float16 v8h;
typedef __attribute__((ext_vector_type(8)))  float    v8f;
typedef __attribute__((ext_vector_type(4)))  float    v4f;

static constexpr int kN    = 36;    // sensors
static constexpr int kDOB  = 4;     // obs embedding
static constexpr int kT    = 60;    // timesteps
static constexpr int kDE   = 160;   // transformer width
static constexpr int kH    = 4;     // heads
static constexpr int kHD   = 40;    // head dim
static constexpr int kNHID = 128;
static constexpr int kB    = 2048;  // batch
static constexpr int kF    = 240;   // per-node feature (T*DOB)
static constexpr int kDF   = 196;   // final mlp width

// ---------------------------------------------------------------------------
// WMMA GEMM: C[M,N] = A[M,K] @ W[N,K]^T + bias (+ residual) (+ relu)
// Block tile 128x64, 8 waves; each wave owns a 32x32 tile (4 WMMAs/chunk).
// ---------------------------------------------------------------------------
__device__ inline v8h zero8() {
  v8h h;
#pragma unroll
  for (int i = 0; i < 8; ++i) h[i] = (_Float16)0.0f;
  return h;
}

__device__ inline v8h cvt8(const float* __restrict__ p) {
  const v4f f0 = *reinterpret_cast<const v4f*>(p);
  const v4f f1 = *reinterpret_cast<const v4f*>(p + 4);
  v8h h;
#pragma unroll
  for (int i = 0; i < 4; ++i) {
    h[i]     = (_Float16)f0[i];
    h[i + 4] = (_Float16)f1[i];
  }
  return h;
}

// Tail-chunk segment load: rem = K - ks (may be <= 0).
__device__ inline v8h load_tail(const float* __restrict__ p, int rem) {
  if (rem >= 8) return cvt8(p);
  v8h h = zero8();
  if (rem > 0) {
    const int nv = rem < 8 ? rem : 8;
    for (int i = 0; i < nv; ++i) h[i] = (_Float16)p[i];
  }
  return h;
}

__device__ inline v16h frag_from_lds(const _Float16* rowbase, int lane) {
  // ISA 7.12.2 16-bit A/B layout: lanes 0-15 hold K {0..7,16..23},
  // lanes 16-31 hold K {8..15,24..31}; two 16-byte LDS loads per lane.
  const int ks = (lane >> 4) * 8;
  v8h lo = *reinterpret_cast<const v8h*>(rowbase + ks);
  v8h hi = *reinterpret_cast<const v8h*>(rowbase + ks + 16);
  v16h f;
#pragma unroll
  for (int i = 0; i < 8; ++i) { f[i] = lo[i]; f[i + 8] = hi[i]; }
  return f;
}

__device__ inline void store_c(float v, int m, int n, int N, const float* bias,
                               const float* residual, float* C, int relu) {
  if (n >= N) return;
  const size_t idx = (size_t)m * N + n;
  if (bias) v += bias[n];
  if (residual) v += residual[idx];
  if (relu) v = fmaxf(v, 0.0f);
  C[idx] = v;
}

__global__ __launch_bounds__(256) void gemm_wmma(
    const float* __restrict__ A, const float* __restrict__ W,
    const float* __restrict__ bias, const float* __restrict__ residual,
    float* __restrict__ C, int M, int N, int K, int relu) {
  __shared__ __align__(16) _Float16 As[2][128 * 32];
  __shared__ __align__(16) _Float16 Bs[2][64 * 32];

  const int tid  = threadIdx.x;
  const int lane = tid & 31;
  const int wave = tid >> 5;
  const int m0 = blockIdx.y * 128;
  const int n0 = blockIdx.x * 64;
  const int wm = (wave >> 1) * 32;  // 0,32,64,96
  const int wn = (wave & 1) * 32;   // 0,32

  v8f c00 = {}, c01 = {}, c10 = {}, c11 = {};

  const int lr = tid >> 2;        // 0..63
  const int lk = (tid & 3) * 8;   // 0,8,16,24
  // B rows beyond N only feed output columns that are never stored; clamping
  // keeps the load in-bounds without any zero-fill or predication.
  const int gnB = n0 + lr < N ? n0 + lr : N - 1;

  const float* pA0 = A + (size_t)(m0 + lr) * K + lk;
  const float* pA1 = A + (size_t)(m0 + lr + 64) * K + lk;
  const float* pW  = W + (size_t)gnB * K + lk;

  const int r = lane & 15;

  auto stage = [&](int bf, v8h a0, v8h a1, v8h b) {
    *reinterpret_cast<v8h*>(&As[bf][lr * 32 + lk]) = a0;
    *reinterpret_cast<v8h*>(&As[bf][(lr + 64) * 32 + lk]) = a1;
    *reinterpret_cast<v8h*>(&Bs[bf][lr * 32 + lk]) = b;
  };
  auto compute = [&](int bf) {
    v16h a0 = frag_from_lds(&As[bf][(wm + r) * 32], lane);
    v16h a1 = frag_from_lds(&As[bf][(wm + 16 + r) * 32], lane);
    v16h b0 = frag_from_lds(&Bs[bf][(wn + r) * 32], lane);
    v16h b1 = frag_from_lds(&Bs[bf][(wn + 16 + r) * 32], lane);
    c00 = __builtin_amdgcn_wmma_f32_16x16x32_f16(false, a0, false, b0,
                                                 (short)0, c00, false, false);
    c01 = __builtin_amdgcn_wmma_f32_16x16x32_f16(false, a0, false, b1,
                                                 (short)0, c01, false, false);
    c10 = __builtin_amdgcn_wmma_f32_16x16x32_f16(false, a1, false, b0,
                                                 (short)0, c10, false, false);
    c11 = __builtin_amdgcn_wmma_f32_16x16x32_f16(false, a1, false, b1,
                                                 (short)0, c11, false, false);
  };

  const int Kfull = K & ~31;  // K >= 128 at every call site

  // Prefetch chunk 0.
  v8h ra0 = cvt8(pA0);
  v8h ra1 = cvt8(pA1);
  v8h rb  = cvt8(pW);
  int buf = 0;
  int k0 = 0;

  // Pipelined main loop: stage chunk i, barrier, prefetch chunk i+1 (global
  // loads in flight under the WMMAs), compute chunk i.  One barrier/chunk:
  // reads of buffer b (iter i) and writes (iter i+2) are separated by the
  // iter i+1 barrier.
  for (; k0 + 32 < Kfull; k0 += 32) {
    stage(buf, ra0, ra1, rb);
    __syncthreads();
    ra0 = cvt8(pA0 + k0 + 32);
    ra1 = cvt8(pA1 + k0 + 32);
    rb  = cvt8(pW + k0 + 32);
    compute(buf);
    buf ^= 1;
  }

  // Last full chunk; prefetch the (optional) tail chunk.
  const bool tail = Kfull < K;  // K = 240 or 196
  stage(buf, ra0, ra1, rb);
  __syncthreads();
  if (tail) {
    const int rem = K - (Kfull + lk);
    ra0 = load_tail(pA0 + Kfull, rem);
    ra1 = load_tail(pA1 + Kfull, rem);
    rb  = load_tail(pW + Kfull, rem);
  }
  compute(buf);
  buf ^= 1;
  if (tail) {
    stage(buf, ra0, ra1, rb);
    __syncthreads();
    compute(buf);
  }

  // C/D layout: VGPR i -> M = i + 8*(lane>=16), N = lane&15
  const int moff = (lane >> 4) * 8;
#pragma unroll
  for (int i = 0; i < 8; ++i) {
    const int mA = m0 + wm + i + moff;
    const int mB = mA + 16;
    const int nA = n0 + wn + r;
    const int nB = nA + 16;
    store_c(c00[i], mA, nA, N, bias, residual, C, relu);
    store_c(c01[i], mA, nB, N, bias, residual, C, relu);
    store_c(c10[i], mB, nA, N, bias, residual, C, relu);
    store_c(c11[i], mB, nB, N, bias, residual, C, relu);
  }
}

// ---------------------------------------------------------------------------
// hg[b,n,t*4+j] = relu(src[t,b,n] * R_u[n*4+j])
// ---------------------------------------------------------------------------
__global__ void build_hg(const float* __restrict__ src,
                         const float* __restrict__ R_u,
                         float* __restrict__ hg, int total) {
  int idx = blockIdx.x * blockDim.x + threadIdx.x;
  if (idx >= total) return;
  const int f = idx % kF;
  const int n = (idx / kF) % kN;
  const int b = idx / (kF * kN);
  const int t = f >> 2, j = f & 3;
  const float s = src[((size_t)t * kB + b) * (2 * kN) + n];
  const float v = s * R_u[n * kDOB + j];
  hg[idx] = v > 0.0f ? v : 0.0f;
}

// vbar[b,f] = mean_n V[b,n,f]
__global__ void vbar_kernel(const float* __restrict__ V,
                            float* __restrict__ vbar, int total) {
  int idx = blockIdx.x * blockDim.x + threadIdx.x;
  if (idx >= total) return;
  const int f = idx % kF;
  const int b = idx / kF;
  float s = 0.0f;
  for (int n = 0; n < kN; ++n) s += V[((size_t)b * kN + n) * kF + f];
  vbar[idx] = s * (1.0f / kN);
}

// beta[b,n] = sigmoid( sum_f Wb[f]*vbar + Wb[F+f]*xr + Wb[2F+f]*(vbar-xr) + bb )
__global__ __launch_bounds__(256) void beta_kernel(
    const float* __restrict__ vbar, const float* __restrict__ xr,
    const float* __restrict__ Wb, const float* __restrict__ bb,
    float* __restrict__ beta) {
  const int row  = blockIdx.x * 8 + (threadIdx.x >> 5);  // b*kN + n
  const int lane = threadIdx.x & 31;
  const int b = row / kN;
  const float* xrp = xr + (size_t)row * kF;
  const float* vbp = vbar + (size_t)b * kF;
  float s = 0.0f;
  for (int f = lane; f < kF; f += 32) {
    const float o = vbp[f], x = xrp[f];
    s += Wb[f] * o + Wb[kF + f] * x + Wb[2 * kF + f] * (o - x);
  }
#pragma unroll
  for (int off = 16; off > 0; off >>= 1) s += __shfl_xor(s, off, 32);
  if (lane == 0) beta[row] = 1.0f / (1.0f + expf(-(s + bb[0])));
}

// h[b,n,f] = beta*xr + (1-beta)*vbar
__global__ void combine_kernel(const float* __restrict__ beta,
                               const float* __restrict__ xr,
                               const float* __restrict__ vbar,
                               float* __restrict__ out, int total) {
  int idx = blockIdx.x * blockDim.x + threadIdx.x;
  if (idx >= total) return;
  const int f = idx % kF;
  const int row = idx / kF;
  const int b = row / kN;
  const float be = beta[row];
  out[idx] = be * xr[idx] + (1.0f - be) * vbar[(size_t)b * kF + f];
}

// emb[b,n] = static[b,:] . W_static[n,:] + b_static[n]
__global__ void emb_kernel(const float* __restrict__ st,
                           const float* __restrict__ Ws,
                           const float* __restrict__ bs,
                           float* __restrict__ emb, int total) {
  int idx = blockIdx.x * blockDim.x + threadIdx.x;
  if (idx >= total) return;
  const int n = idx % kN, b = idx / kN;
  float s = bs[n];
  for (int k = 0; k < 9; ++k) s += st[b * 9 + k] * Ws[n * 9 + k];
  emb[idx] = s;
}

// xT[t,b,:144] from h2, xT[t,b,144:160] = positional encoding
__global__ void build_xt(const float* __restrict__ h2,
                         const float* __restrict__ times,
                         float* __restrict__ xT, int total) {
  int idx = blockIdx.x * blockDim.x + threadIdx.x;
  if (idx >= total) return;
  const int d = idx % kDE;
  const int b = (idx / kDE) % kB;
  const int t = idx / (kDE * kB);
  float v;
  if (d < kN * kDOB) {
    const int n = d >> 2, j = d & 3;
    v = h2[((size_t)b * kN + n) * kF + t * kDOB + j];
  } else {
    const int i = d - kN * kDOB;
    const int k = (i < 8) ? i : i - 8;
    const float ts = powf((float)kT, (float)k * (1.0f / 7.0f));
    const float a = times[(size_t)t * kB + b] / ts;
    v = (i < 8) ? sinf(a) : cosf(a);
  }
  xT[idx] = v;
}

// Per-(b,h) masked attention: qkv (T,B,480) -> o (T,B,160)
__global__ __launch_bounds__(64) void attn_kernel(
    const float* __restrict__ qkv, const int* __restrict__ lengths,
    float* __restrict__ o) {
  __shared__ float qs[kT * kHD];
  __shared__ float ks_[kT * kHD];
  __shared__ float vs[kT * kHD];
  __shared__ float ps[kT * kT];
  const int b = blockIdx.x / kH;
  const int h = blockIdx.x % kH;
  const int len = lengths[b];
  for (int i = threadIdx.x; i < kT * kHD; i += 64) {
    const int t = i / kHD, d = i % kHD;
    const size_t base = ((size_t)t * kB + b) * (3 * kDE) + h * kHD + d;
    qs[i]  = qkv[base];
    ks_[i] = qkv[base + kDE];
    vs[i]  = qkv[base + 2 * kDE];
  }
  __syncthreads();
  const int t = threadIdx.x;
  if (t < kT) {
    const float scale = 0.158113883f;  // 1/sqrt(40)
    float mx = -1e30f;
    for (int s = 0; s < kT; ++s) {
      float acc = 0.0f;
      for (int d = 0; d < kHD; ++d) acc += qs[t * kHD + d] * ks_[s * kHD + d];
      acc = (s < len) ? acc * scale : -1e9f;
      ps[t * kT + s] = acc;
      mx = fmaxf(mx, acc);
    }
    float sum = 0.0f;
    for (int s = 0; s < kT; ++s) {
      const float e = expf(ps[t * kT + s] - mx);
      ps[t * kT + s] = e;
      sum += e;
    }
    const float inv = 1.0f / sum;
    for (int d = 0; d < kHD; ++d) {
      float acc = 0.0f;
      for (int s = 0; s < kT; ++s) acc += ps[t * kT + s] * vs[s * kHD + d];
      o[((size_t)t * kB + b) * kDE + h * kHD + d] = acc * inv;
    }
  }
}

// wave-per-row LayerNorm over kDE=160
__global__ __launch_bounds__(256) void ln_kernel(
    const float* __restrict__ x, const float* __restrict__ g,
    const float* __restrict__ bt, float* __restrict__ y, int rows) {
  const int row = blockIdx.x * 8 + (threadIdx.x >> 5);
  if (row >= rows) return;
  const int lane = threadIdx.x & 31;
  const float* xp = x + (size_t)row * kDE;
  float vals[5];
  float s = 0.0f;
#pragma unroll
  for (int i = 0; i < 5; ++i) { vals[i] = xp[lane + 32 * i]; s += vals[i]; }
#pragma unroll
  for (int off = 16; off > 0; off >>= 1) s += __shfl_xor(s, off, 32);
  const float mean = s * (1.0f / kDE);
  float v = 0.0f;
#pragma unroll
  for (int i = 0; i < 5; ++i) { const float d = vals[i] - mean; v += d * d; }
#pragma unroll
  for (int off = 16; off > 0; off >>= 1) v += __shfl_xor(v, off, 32);
  const float inv = rsqrtf(v * (1.0f / kDE) + 1e-5f);
#pragma unroll
  for (int i = 0; i < 5; ++i) {
    const int c = lane + 32 * i;
    y[(size_t)row * kDE + c] = (vals[i] - mean) * inv * g[c] + bt[c];
  }
}

// agg[b,d] = sum_{t<len} xT[t,b,d] / (len+1)
__global__ void agg_kernel(const float* __restrict__ xT,
                           const int* __restrict__ lengths,
                           float* __restrict__ agg, int total) {
  int idx = blockIdx.x * blockDim.x + threadIdx.x;
  if (idx >= total) return;
  const int d = idx % kDE, b = idx / kDE;
  const int len = lengths[b];
  float s = 0.0f;
  for (int t = 0; t < kT; ++t)
    if (t < len) s += xT[((size_t)t * kB + b) * kDE + d];
  agg[idx] = s / ((float)len + 1.0f);
}

__global__ void feat_kernel(const float* __restrict__ agg,
                            const float* __restrict__ emb,
                            float* __restrict__ feat, int total) {
  int idx = blockIdx.x * blockDim.x + threadIdx.x;
  if (idx >= total) return;
  const int j = idx % kDF, b = idx / kDF;
  feat[idx] = (j < kDE) ? agg[(size_t)b * kDE + j] : emb[(size_t)b * kN + (j - kDE)];
}

// logits = hid @ w2.T + b2 ; out[B*2] = dist (exactly 0)
__global__ void final_kernel(const float* __restrict__ hid,
                             const float* __restrict__ w2,
                             const float* __restrict__ b2,
                             float* __restrict__ out) {
  int idx = blockIdx.x * blockDim.x + threadIdx.x;
  if (idx == kB * 2) { out[idx] = 0.0f; return; }
  if (idx > kB * 2) return;
  const int b = idx >> 1, c = idx & 1;
  float s = b2[c];
  const float* hp = hid + (size_t)b * kDF;
  const float* wp = w2 + c * kDF;
  for (int k = 0; k < kDF; ++k) s += hp[k] * wp[k];
  out[idx] = s;
}

// ---------------------------------------------------------------------------
extern "C" void kernel_launch(void* const* d_in, const int* in_sizes, int n_in,
                              void* d_out, int out_size, void* d_ws,
                              size_t ws_size, hipStream_t stream) {
  (void)in_sizes; (void)n_in; (void)out_size; (void)ws_size;
  const float* src      = (const float*)d_in[0];
  const float* stat     = (const float*)d_in[1];
  const float* times    = (const float*)d_in[2];
  const int*   lengths  = (const int*)d_in[3];
  const float* R_u      = (const float*)d_in[4];
  const float* W_static = (const float*)d_in[5];
  const float* b_static = (const float*)d_in[6];
  const float* opWv[2]  = {(const float*)d_in[7],  (const float*)d_in[13]};
  const float* opbv[2]  = {(const float*)d_in[8],  (const float*)d_in[14]};
  const float* opWs[2]  = {(const float*)d_in[9],  (const float*)d_in[15]};
  const float* opbs[2]  = {(const float*)d_in[10], (const float*)d_in[16]};
  const float* opWb[2]  = {(const float*)d_in[11], (const float*)d_in[17]};
  const float* opbb[2]  = {(const float*)d_in[12], (const float*)d_in[18]};
  const float* in_proj_w  = (const float*)d_in[19];
  const float* in_proj_b  = (const float*)d_in[20];
  const float* out_proj_w = (const float*)d_in[21];
  const float* out_proj_b = (const float*)d_in[22];
  const float* lin1_w = (const float*)d_in[23];
  const float* lin1_b = (const float*)d_in[24];
  const float* lin2_w = (const float*)d_in[25];
  const float* lin2_b = (const float*)d_in[26];
  const float* ln1_g = (const float*)d_in[27];
  const float* ln1_b = (const float*)d_in[28];
  const float* ln2_g = (const float*)d_in[29];
  const float* ln2_b = (const float*)d_in[30];
  const float* mlp_w1 = (const float*)d_in[31];
  const float* mlp_b1 = (const float*)d_in[32];
  const float* mlp_w2 = (const float*)d_in[33];
  const float* mlp_b2 = (const float*)d_in[34];

  float* ws = (float*)d_ws;
  const size_t SEG = (size_t)kB * kN * kF;          // 17,694,720
  const size_t ROW = (size_t)kT * kB;               // 122,880
  float* hg  = ws;                                  // graph region (4 SEG)
  float* h1  = ws + SEG;
  float* V   = ws + 2 * SEG;
  float* XR  = ws + 3 * SEG;
  float* qkv = ws;                                  // overlays graph region
  float* xT  = ws + 4 * SEG;                        // T*B*160
  float* o   = xT + ROW * kDE;                      // T*B*160
  float* tmp = o + ROW * kDE;                       // T*B*160
  float* ffh = tmp + ROW * kDE;                     // T*B*128
  float* vbar = ffh + ROW * kNHID;                  // B*240
  float* beta = vbar + (size_t)kB * kF;             // B*36
  float* emb  = beta + (size_t)kB * kN;             // B*36
  float* agg  = emb + (size_t)kB * kN;              // B*160
  float* feat = agg + (size_t)kB * kDE;             // B*196
  float* hid  = feat + (size_t)kB * kDF;            // B*196

  const int TPB = 256;
  auto ceil_div = [](long long a, long long b) { return (int)((a + b - 1) / b); };

  // 1) node features + static embedding
  build_hg<<<ceil_div(SEG, TPB), TPB, 0, stream>>>(src, R_u, hg, (int)SEG);
  emb_kernel<<<ceil_div(kB * kN, TPB), TPB, 0, stream>>>(stat, W_static,
                                                         b_static, emb, kB * kN);

  // 2) two graph-propagation layers (uniform alpha -> node-mean aggregation)
  const float* gin[2]  = {hg, h1};
  float*       gout[2] = {h1, hg};
  const int MG = kB * kN;  // 73728 = 128*576
  dim3 gg(ceil_div(kF, 64), MG / 128);
  for (int l = 0; l < 2; ++l) {
    gemm_wmma<<<gg, 256, 0, stream>>>(gin[l], opWv[l], opbv[l], nullptr, V,
                                      MG, kF, kF, 0);
    gemm_wmma<<<gg, 256, 0, stream>>>(gin[l], opWs[l], opbs[l], nullptr, XR,
                                      MG, kF, kF, 0);
    vbar_kernel<<<ceil_div((long long)kB * kF, TPB), TPB, 0, stream>>>(
        V, vbar, kB * kF);
    beta_kernel<<<MG / 8, 256, 0, stream>>>(vbar, XR, opWb[l], opbb[l], beta);
    combine_kernel<<<ceil_div(SEG, TPB), TPB, 0, stream>>>(beta, XR, vbar,
                                                           gout[l], (int)SEG);
  }

  // 3) transformer input (h2 lives in hg) + positional encoding
  build_xt<<<ceil_div(ROW * kDE, TPB), TPB, 0, stream>>>(hg, times, xT,
                                                         (int)(ROW * kDE));

  // 4) two post-LN transformer layers
  const int MT = (int)ROW;  // 122880 = 128*960
  dim3 gqkv(ceil_div(3 * kDE, 64), MT / 128);
  dim3 gprj(ceil_div(kDE, 64), MT / 128);
  dim3 gff1(ceil_div(kNHID, 64), MT / 128);
  for (int l = 0; l < 2; ++l) {
    gemm_wmma<<<gqkv, 256, 0, stream>>>(xT, in_proj_w + (size_t)l * 3 * kDE * kDE,
                                        in_proj_b + l * 3 * kDE, nullptr, qkv,
                                        MT, 3 * kDE, kDE, 0);
    attn_kernel<<<kB * kH, 64, 0, stream>>>(qkv, lengths, o);
    gemm_wmma<<<gprj, 256, 0, stream>>>(o, out_proj_w + (size_t)l * kDE * kDE,
                                        out_proj_b + l * kDE, xT, tmp,
                                        MT, kDE, kDE, 0);
    ln_kernel<<<MT / 8, 256, 0, stream>>>(tmp, ln1_g + l * kDE, ln1_b + l * kDE,
                                          xT, MT);
    gemm_wmma<<<gff1, 256, 0, stream>>>(xT, lin1_w + (size_t)l * kNHID * kDE,
                                        lin1_b + l * kNHID, nullptr, ffh,
                                        MT, kNHID, kDE, 1);
    gemm_wmma<<<gprj, 256, 0, stream>>>(ffh, lin2_w + (size_t)l * kDE * kNHID,
                                        lin2_b + l * kDE, xT, tmp,
                                        MT, kDE, kNHID, 0);
    ln_kernel<<<MT / 8, 256, 0, stream>>>(tmp, ln2_g + l * kDE, ln2_b + l * kDE,
                                          xT, MT);
  }

  // 5) masked time aggregation, concat, classifier head
  agg_kernel<<<ceil_div(kB * kDE, TPB), TPB, 0, stream>>>(xT, lengths, agg,
                                                          kB * kDE);
  feat_kernel<<<ceil_div(kB * kDF, TPB), TPB, 0, stream>>>(agg, emb, feat,
                                                           kB * kDF);
  dim3 gmlp(ceil_div(kDF, 64), kB / 128);
  gemm_wmma<<<gmlp, 256, 0, stream>>>(feat, mlp_w1, mlp_b1, nullptr, hid,
                                      kB, kDF, kDF, 1);
  final_kernel<<<ceil_div(kB * 2 + 1, TPB), TPB, 0, stream>>>(
      hid, mlp_w2, mlp_b2, (float*)d_out);
}
